// Decoder_49615462203898
// MI455X (gfx1250) — compile-verified, matrix operand order checked by reading, not run on hardware
//
#include <hip/hip_runtime.h>
#include <stdint.h>

// Problem constants (match reference)
#define S_STEPS 12
#define NN      50000
#define DIN     64
#define DHD     128
#define CC      2
#define EE      600000
#define HIDN    65
#define KDIM    576      // 3*64 (x-cheb) + 3*128 (h-cheb)
#define GDIM    512      // 4 gates * 128
#define KT      18       // KDIM / 32

typedef __bf16          v16bf __attribute__((ext_vector_type(16)));
typedef unsigned short  v16u  __attribute__((ext_vector_type(16)));
typedef float           v8f   __attribute__((ext_vector_type(8)));

__device__ __forceinline__ unsigned short f32_to_bf16(float f) {
  union { float f; unsigned u; } v; v.f = f;
  unsigned u = v.u;
  if ((u & 0x7fffffffu) > 0x7f800000u) return (unsigned short)((u >> 16) | 0x40); // NaN
  u += 0x7fffu + ((u >> 16) & 1u);   // round-to-nearest-even
  return (unsigned short)(u >> 16);
}

// ---------------- generic fill ----------------
__global__ void k_fill(float* __restrict__ p, float v, long long n) {
  long long i = (long long)blockIdx.x * blockDim.x + threadIdx.x;
  if (i < n) p[i] = v;
}

// ---------------- graph normalization ----------------
__global__ void k_degree(const long long* __restrict__ src, float* __restrict__ deg, int nE) {
  int e = blockIdx.x * blockDim.x + threadIdx.x;
  if (e < nE) atomicAdd(&deg[(int)src[e]], 1.0f);
}
__global__ void k_dinv(const float* __restrict__ deg, float* __restrict__ dinv, int n) {
  int i = blockIdx.x * blockDim.x + threadIdx.x;
  if (i < n) { float d = deg[i]; dinv[i] = (d > 0.f) ? rsqrtf(fmaxf(d, 1.f)) : 0.f; }
}
__global__ void k_norm(const long long* __restrict__ src, const long long* __restrict__ dst,
                       const float* __restrict__ dinv, float* __restrict__ nrm, int nE) {
  int e = blockIdx.x * blockDim.x + threadIdx.x;
  if (e < nE) nrm[e] = -dinv[(int)src[e]] * dinv[(int)dst[e]];
}

// ---------------- SpMM scatter: Y[dst] += norm * X[src]  (dim = 1<<dimShift) ----------------
__global__ void k_spmm(const long long* __restrict__ src, const long long* __restrict__ dst,
                       const float* __restrict__ nrm, const float* __restrict__ X,
                       float* __restrict__ Y, int dimShift, long long total) {
  long long i = (long long)blockIdx.x * blockDim.x + threadIdx.x;
  if (i >= total) return;
  int       d = (int)(i & ((1 << dimShift) - 1));
  long long e = i >> dimShift;
  int s = (int)src[e], t = (int)dst[e];
  atomicAdd(&Y[((size_t)t << dimShift) + d], nrm[e] * X[((size_t)s << dimShift) + d]);
}

// Cheb recurrence finish: Y = 2*Y - X
__global__ void k_cheb2(float* __restrict__ Y, const float* __restrict__ X, long long n) {
  long long i = (long long)blockIdx.x * blockDim.x + threadIdx.x;
  if (i < n) Y[i] = 2.f * Y[i] - X[i];
}

// ---------------- pack weights into WMMA-fragment-native bf16 layout ----------------
// Bp[((kt*512 + c)*32) + e]: per (k-tile, column), 32 bf16 K-values ordered so each
// lane's 16-element fragment is one contiguous 32B read (ISA 16-bit B layout).
__global__ void k_packB(const float* __restrict__ Wx, const float* __restrict__ Wh,
                        unsigned short* __restrict__ Bp) {
  int idx = blockIdx.x * blockDim.x + threadIdx.x;
  if (idx >= KT * GDIM * 32) return;
  int e    = idx & 31;
  int c    = (idx >> 5) & (GDIM - 1);
  int kt   = idx >> 14;                 // / (512*32)
  int half = e >> 4;                    // 0: lanes 0-15, 1: lanes 16-31
  int e2   = e & 15;
  int kl   = (e2 < 8) ? ((half ? 8 : 0) + e2) : (16 + (half ? 8 : 0) + (e2 - 8));
  int K    = kt * 32 + kl;
  int g    = c >> 7;                    // gate 0..3 (i,f,c,o)
  int dh   = c & 127;
  float v;
  if (K < 3 * DIN) {                    // x-cheb block: Wx[4,3,64,128]
    int k = K / DIN, din = K % DIN;
    v = Wx[((size_t)(g * 3 + k) * DIN + din) * DHD + dh];
  } else {                              // h-cheb block: Wh[4,3,128,128]
    int K2 = K - 3 * DIN;
    int k = K2 >> 7, di = K2 & 127;
    v = Wh[((size_t)(g * 3 + k) * DHD + di) * DHD + dh];
  }
  Bp[idx] = f32_to_bf16(v);
}

__global__ void k_bias(const float* __restrict__ bx, const float* __restrict__ bh,
                       const float* __restrict__ bg, float* __restrict__ bias) {
  int j = blockIdx.x * blockDim.x + threadIdx.x;
  if (j < GDIM) bias[j] = bx[j] + bh[j] + bg[j];
}

// ---------------- build bf16 A = [T0x|T1x|T2x|T0h|T1h|T2h]  [N,576] ----------------
__global__ void k_buildA(const float* __restrict__ Xt,  const float* __restrict__ T1x,
                         const float* __restrict__ T2x, const float* __restrict__ H0,
                         const float* __restrict__ T1h, const float* __restrict__ T2h,
                         unsigned short* __restrict__ Abf) {
  long long i = (long long)blockIdx.x * blockDim.x + threadIdx.x;
  if (i >= (long long)NN * KDIM) return;
  int j = (int)(i % KDIM);
  int n = (int)(i / KDIM);
  float v;
  if      (j < 64)  v = Xt [(size_t)n * 64  + j];
  else if (j < 128) v = T1x[(size_t)n * 64  + j - 64];
  else if (j < 192) v = T2x[(size_t)n * 64  + j - 128];
  else if (j < 320) v = H0 [(size_t)n * 128 + j - 192];
  else if (j < 448) v = T1h[(size_t)n * 128 + j - 320];
  else              v = T2h[(size_t)n * 128 + j - 448];
  Abf[i] = f32_to_bf16(v);
}

// ---------------- WMMA GEMM: G[N,512] = A[N,576] x B[576,512] + bias ----------------
__device__ __forceinline__ v16bf load_a_frag(const unsigned short* __restrict__ A,
                                             int row0, int k0, int lane) {
  int r = row0 + (lane & 15);
  const unsigned short* p = A + (size_t)r * KDIM + k0 + ((lane & 16) ? 8 : 0);
  v16u raw;
#pragma unroll
  for (int e = 0; e < 8; ++e) raw[e] = p[e];          // K = khalf + e
#pragma unroll
  for (int e = 0; e < 8; ++e) raw[8 + e] = p[16 + e]; // K = 16 + khalf + e
  return __builtin_bit_cast(v16bf, raw);
}
__device__ __forceinline__ v16bf load_b_frag(const unsigned short* __restrict__ Bp,
                                             int kt, int col, int lane) {
  const unsigned short* p =
      Bp + (((size_t)kt * GDIM + (col + (lane & 15))) << 5) + ((lane & 16) ? 16 : 0);
  v16u raw;
#pragma unroll
  for (int e = 0; e < 16; ++e) raw[e] = p[e];         // contiguous 32B
  return __builtin_bit_cast(v16bf, raw);
}

__global__ void __launch_bounds__(128)
k_gemm(const unsigned short* __restrict__ Abf, const unsigned short* __restrict__ Bp,
       const float* __restrict__ bias, float* __restrict__ G) {
  const int lane = threadIdx.x & 31;
  const int wave = threadIdx.x >> 5;
  const int row0 = (blockIdx.x >> 1) * 16;                    // N = 3125*16 exactly
  const int col0 = (((blockIdx.x & 1) << 2) | wave) * 64;     // 8 col-groups of 64
  v8f acc0 = {}, acc1 = {}, acc2 = {}, acc3 = {};
  for (int kt = 0; kt < KT; ++kt) {
    v16bf a  = load_a_frag(Abf, row0, kt * 32, lane);
    v16bf b0 = load_b_frag(Bp, kt, col0,      lane);
    v16bf b1 = load_b_frag(Bp, kt, col0 + 16, lane);
    v16bf b2 = load_b_frag(Bp, kt, col0 + 32, lane);
    v16bf b3 = load_b_frag(Bp, kt, col0 + 48, lane);
    acc0 = __builtin_amdgcn_wmma_f32_16x16x32_bf16(false, a, false, b0, (short)0, acc0, false, false);
    acc1 = __builtin_amdgcn_wmma_f32_16x16x32_bf16(false, a, false, b1, (short)0, acc1, false, false);
    acc2 = __builtin_amdgcn_wmma_f32_16x16x32_bf16(false, a, false, b2, (short)0, acc2, false, false);
    acc3 = __builtin_amdgcn_wmma_f32_16x16x32_bf16(false, a, false, b3, (short)0, acc3, false, false);
  }
  const int c0 = col0 + (lane & 15);
  const int rb = row0 + ((lane & 16) ? 8 : 0);
  const float bi0 = bias[c0], bi1 = bias[c0 + 16], bi2 = bias[c0 + 32], bi3 = bias[c0 + 48];
#pragma unroll
  for (int v = 0; v < 8; ++v) {
    size_t base = (size_t)(rb + v) * GDIM;
    G[base + c0     ] = acc0[v] + bi0;
    G[base + c0 + 16] = acc1[v] + bi1;
    G[base + c0 + 32] = acc2[v] + bi2;
    G[base + c0 + 48] = acc3[v] + bi3;
  }
}

// ---------------- LSTM pointwise (peephole) ----------------
__global__ void k_lstm(const float* __restrict__ G, const float* __restrict__ wc,
                       float* __restrict__ h, float* __restrict__ c) {
  int i = blockIdx.x * blockDim.x + threadIdx.x;
  if (i >= NN * DHD) return;
  int n = i >> 7, d = i & 127;
  const float* g = G + (size_t)n * GDIM;
  float cold = c[i];
  float iv = 1.f / (1.f + expf(-(g[d]       + wc[d]       * cold)));
  float fv = 1.f / (1.f + expf(-(g[128 + d] + wc[128 + d] * cold)));
  float tv = tanhf(g[256 + d]);
  float cn = fv * cold + iv * tv;
  float ov = 1.f / (1.f + expf(-(g[384 + d] + wc[256 + d] * cn)));
  h[i] = ov * tanhf(cn);
  c[i] = cn;
}

// ---------------- fused head: tanh -> LN -> MLP -> sigmoid, per step ----------------
__global__ void __launch_bounds__(256)
k_head(const float* __restrict__ h, const float* __restrict__ lng, const float* __restrict__ lnb,
       const float* __restrict__ W1, const float* __restrict__ b1,
       const float* __restrict__ W2, const float* __restrict__ b2,
       float* __restrict__ out, int t) {
  __shared__ float xln[8][DHD + 8];
  __shared__ float y1s[8][HIDN + 7];
  const int lane = threadIdx.x & 31, wave = threadIdx.x >> 5;
  const int n = blockIdx.x * 8 + wave;                 // NN divisible by 8
  float xv[4]; float s = 0.f;
#pragma unroll
  for (int i = 0; i < 4; ++i) { xv[i] = tanhf(h[(size_t)n * DHD + lane + i * 32]); s += xv[i]; }
#pragma unroll
  for (int off = 16; off; off >>= 1) s += __shfl_xor(s, off, 32);
  const float mu = s * (1.0f / DHD);
  float vs = 0.f;
#pragma unroll
  for (int i = 0; i < 4; ++i) { float d = xv[i] - mu; vs += d * d; }
#pragma unroll
  for (int off = 16; off; off >>= 1) vs += __shfl_xor(vs, off, 32);
  const float rstd = rsqrtf(vs * (1.0f / DHD) + 1e-5f);
#pragma unroll
  for (int i = 0; i < 4; ++i) {
    int d = lane + i * 32;
    xln[wave][d] = (xv[i] - mu) * rstd * lng[d] + lnb[d];
  }
  __syncthreads();
  {
    const int j0 = lane, j1 = lane + 32;
    float a0 = b1[j0], a1 = b1[j1];
    for (int d = 0; d < DHD; ++d) {
      float xd = xln[wave][d];
      a0 += xd * W1[d * HIDN + j0];
      a1 += xd * W1[d * HIDN + j1];
    }
    y1s[wave][j0] = fmaxf(a0, 0.f);
    y1s[wave][j1] = fmaxf(a1, 0.f);
    if (lane == 0) {
      float a2 = b1[64];
      for (int d = 0; d < DHD; ++d) a2 += xln[wave][d] * W1[d * HIDN + 64];
      y1s[wave][64] = fmaxf(a2, 0.f);
    }
  }
  __syncthreads();
  if (lane < CC) {
    float a = b2[lane];
    for (int j = 0; j < HIDN; ++j) a += y1s[wave][j] * W2[j * CC + lane];
    out[((size_t)n * CC + lane) * S_STEPS + t] = 1.f / (1.f + expf(-a));
  }
}

// ======================= host =======================
extern "C" void kernel_launch(void* const* d_in, const int* in_sizes, int n_in,
                              void* d_out, int out_size, void* d_ws, size_t ws_size,
                              hipStream_t stream) {
  const float*     H   = (const float*)d_in[0];
  const long long* ei  = (const long long*)d_in[1];   // int64 [2,E]
  const long long* src = ei;
  const long long* dst = ei + EE;
  const float* Wx  = (const float*)d_in[2];
  const float* bx  = (const float*)d_in[3];
  const float* Wh  = (const float*)d_in[4];
  const float* bh  = (const float*)d_in[5];
  const float* bg  = (const float*)d_in[6];
  const float* wc  = (const float*)d_in[7];
  const float* lng = (const float*)d_in[8];
  const float* lnb = (const float*)d_in[9];
  const float* W1  = (const float*)d_in[10];
  const float* b1  = (const float*)d_in[11];
  const float* W2  = (const float*)d_in[12];
  const float* b2  = (const float*)d_in[13];
  float* out = (float*)d_out;

  // ---- workspace carve-up (all sizes 256B-aligned so T1x..T2h stay contiguous) ----
  char* ws = (char*)d_ws; size_t off = 0;
  auto carve = [&](size_t bytes) {
    char* p = ws + off; off += (bytes + 255) & ~(size_t)255; return p;
  };
  float* deg  = (float*)carve((size_t)NN * 4);
  float* dinv = (float*)carve((size_t)NN * 4);
  float* nrm  = (float*)carve((size_t)EE * 4);
  float* T1x  = (float*)carve((size_t)NN * 64  * 4);
  float* T2x  = (float*)carve((size_t)NN * 64  * 4);
  float* T1h  = (float*)carve((size_t)NN * 128 * 4);
  float* T2h  = (float*)carve((size_t)NN * 128 * 4);
  float* hbuf = (float*)carve((size_t)NN * 128 * 4);
  float* cbuf = (float*)carve((size_t)NN * 128 * 4);
  float* G    = (float*)carve((size_t)NN * GDIM * 4);
  unsigned short* Abf = (unsigned short*)carve((size_t)NN * KDIM * 2);
  unsigned short* Bp  = (unsigned short*)carve((size_t)KT * GDIM * 32 * 2);
  float* bias = (float*)carve((size_t)GDIM * 4);

  const int TB = 256;
  auto g1 = [](long long n) { return (unsigned)((n + 255) / 256); };

  // ---- setup: graph norm, weight pack, bias fold, zero states ----
  k_fill  <<<g1(NN), TB, 0, stream>>>(deg, 0.f, NN);
  k_degree<<<g1(EE), TB, 0, stream>>>(src, deg, EE);
  k_dinv  <<<g1(NN), TB, 0, stream>>>(deg, dinv, NN);
  k_norm  <<<g1(EE), TB, 0, stream>>>(src, dst, dinv, nrm, EE);
  k_packB <<<g1((long long)KT * GDIM * 32), TB, 0, stream>>>(Wx, Wh, Bp);
  k_bias  <<<g1(GDIM), TB, 0, stream>>>(bx, bh, bg, bias);
  k_fill  <<<g1((long long)NN * 256), TB, 0, stream>>>(hbuf, 0.f, (long long)NN * 256); // h & c (contiguous)

  const long long totX = (long long)EE * 64;
  const long long totH = (long long)EE * 128;

  for (int t = 0; t < S_STEPS; ++t) {
    const float* Xt = H + (size_t)t * NN * DIN;
    // zero T1x,T2x,T1h,T2h in one shot (contiguous, N*384 floats)
    k_fill <<<g1((long long)NN * 384), TB, 0, stream>>>(T1x, 0.f, (long long)NN * 384);
    // Chebyshev bases for x_t (dim 64)
    k_spmm <<<g1(totX), TB, 0, stream>>>(src, dst, nrm, Xt,  T1x, 6, totX);
    k_spmm <<<g1(totX), TB, 0, stream>>>(src, dst, nrm, T1x, T2x, 6, totX);
    k_cheb2<<<g1((long long)NN * 64), TB, 0, stream>>>(T2x, Xt, (long long)NN * 64);
    // Chebyshev bases for h (dim 128)
    k_spmm <<<g1(totH), TB, 0, stream>>>(src, dst, nrm, hbuf, T1h, 7, totH);
    k_spmm <<<g1(totH), TB, 0, stream>>>(src, dst, nrm, T1h,  T2h, 7, totH);
    k_cheb2<<<g1((long long)NN * 128), TB, 0, stream>>>(T2h, hbuf, (long long)NN * 128);
    // one fused bf16 GEMM for all 4 gates (x and h paths together)
    k_buildA<<<g1((long long)NN * KDIM), TB, 0, stream>>>(Xt, T1x, T2x, hbuf, T1h, T2h, Abf);
    k_gemm  <<<(NN / 16) * 2, 128, 0, stream>>>(Abf, Bp, bias, G);
    // LSTM pointwise + fused head for this step
    k_lstm  <<<g1((long long)NN * DHD), TB, 0, stream>>>(G, wc, hbuf, cbuf);
    k_head  <<<NN / 8, 256, 0, stream>>>(hbuf, lng, lnb, W1, b1, W2, b2, out, t);
  }
}
